// GAT_76124000354827
// MI455X (gfx1250) — matrix operand order repaired and checked by reading, not compiled, not run on hardware
//
#include <hip/hip_runtime.h>
#include <math.h>

typedef __attribute__((ext_vector_type(16))) _Float16 v16h;
typedef __attribute__((ext_vector_type(8)))  _Float16 v8h;
typedef __attribute__((ext_vector_type(4)))  _Float16 v4h;
typedef __attribute__((ext_vector_type(8)))  float    v8f;

#define NNODES 3000
#define NPAD   3008      // 94 * 32
#define INDIM  512
#define HID    64
#define HEADS  8
#define F1     512       // HID*HEADS
#define NCLS   16
#define GALPHA 0.2f

__device__ __forceinline__ float eluf(float x) {
    return x > 0.f ? x : (__expf(x) - 1.f);
}

__device__ __forceinline__ v16h joinh(v8h a, v8h b) {
    return __builtin_shufflevector(a, b, 0, 1, 2, 3, 4, 5, 6, 7,
                                   8, 9, 10, 11, 12, 13, 14, 15);
}

// ---------------- degree / dinv ----------------
__global__ void k_degree(const float* __restrict__ A, float* __restrict__ dinv) {
    int i = blockIdx.x;
    __shared__ float red[256];
    float s = 0.f;
    if (i < NNODES)
        for (int j = threadIdx.x; j < NNODES; j += 256) s += A[(size_t)i * NNODES + j];
    red[threadIdx.x] = s;
    __syncthreads();
    for (int st = 128; st > 0; st >>= 1) {
        if (threadIdx.x < st) red[threadIdx.x] += red[threadIdx.x + st];
        __syncthreads();
    }
    if (threadIdx.x == 0) dinv[i] = (i < NNODES) ? rsqrtf(red[0] + 1.0f) : 0.f;
}

// -------- weight repack: B-fragment-packed f16 layout --------
// frag[(kt*nT + nt)*512 + lane*16 + e] = B[kt*32 + (lane>>4)*16 + e][nt*16 + (lane&15)]
__global__ void k_wcat(const float* __restrict__ Wh, _Float16* __restrict__ Bf) {
    int idx = blockIdx.x * 256 + threadIdx.x;
    if (idx >= (INDIM / 32) * (F1 / 16) * 512) return;
    int e = idx & 15, lane = (idx >> 4) & 31, tile = idx >> 9;
    int kt = tile / (F1 / 16), nt = tile % (F1 / 16);
    int k = kt * 32 + ((lane >> 4) << 4) + e;
    int c = nt * 16 + (lane & 15);
    int h = c >> 6, f = c & 63;
    Bf[idx] = (_Float16)Wh[(h * INDIM + k) * HID + f];
}

__global__ void k_wout(const float* __restrict__ W, _Float16* __restrict__ Bf) {
    int idx = blockIdx.x * 256 + threadIdx.x;
    if (idx >= (INDIM / 32) * 512) return;
    int e = idx & 15, lane = (idx >> 4) & 31, kt = idx >> 9;
    int k = kt * 32 + ((lane >> 4) << 4) + e;
    int c = lane & 15;
    Bf[idx] = (_Float16)W[k * NCLS + c];
}

// ---------------- WMMA GEMM: C[M,N] = A(f32)[M,K] * Bfrag(f16) ----------------
// 256 thr = 8 waves; tile 128 x BN; K step 32; B fragments read straight from global.
// A staging is branch-free: row index clamped (padded rows feed only padded D rows,
// which are stored as literal zeros / not stored at all).
template <int BN>
__global__ void k_gemm(const float* __restrict__ A, int lda,
                       const _Float16* __restrict__ Bf,
                       float* __restrict__ C, _Float16* __restrict__ Chf, int ldc,
                       int M, int N, int K) {
    constexpr int AS = 40;                       // LDS stride in halves (80B rows)
    __shared__ alignas(16) _Float16 As[128 * AS];
    int tid = threadIdx.x, wave = tid >> 5, lane = tid & 31;
    int row0 = blockIdx.x * 128;
    int col0 = blockIdx.y * BN;
    int nT = N >> 4;

    v8f acc[BN / 16];
#pragma unroll
    for (int t = 0; t < BN / 16; ++t) acc[t] = (v8f)0.f;

    for (int k0 = 0; k0 < K; k0 += 32) {
#pragma unroll
        for (int u = 0; u < 4; ++u) {            // stage A 128x32: float4 load -> b64 LDS store
            int q = tid + u * 256;
            int r = q >> 3, kk = (q & 7) * 4;
            int gr = min(row0 + r, M - 1);       // clamp: branch-free, padded rows unused
            float4 v = *(const float4*)(A + (size_t)gr * lda + k0 + kk);
            v4h hv;
            hv[0] = (_Float16)v.x; hv[1] = (_Float16)v.y;
            hv[2] = (_Float16)v.z; hv[3] = (_Float16)v.w;
            *(v4h*)&As[r * AS + kk] = hv;
        }
        __syncthreads();

        int abase = (wave * 16 + (lane & 15)) * AS + ((lane >> 4) << 4);
        v16h af = joinh(*(const v8h*)&As[abase], *(const v8h*)&As[abase + 8]);

#pragma unroll
        for (int t = 0; t < BN / 16; ++t) {
            const _Float16* bp =
                Bf + ((size_t)(k0 >> 5) * nT + (col0 >> 4) + t) * 512 + lane * 16;
            v16h bf = joinh(*(const v8h*)bp, *(const v8h*)(bp + 8));
            acc[t] = __builtin_amdgcn_wmma_f32_16x16x32_f16(
                false, af, false, bf, (short)0, acc[t], false, false);
        }
        __syncthreads();
    }

    int crow = row0 + wave * 16 + (lane >> 4) * 8;
#pragma unroll
    for (int t = 0; t < BN / 16; ++t) {
        int ccol = col0 + t * 16 + (lane & 15);
#pragma unroll
        for (int v = 0; v < 8; ++v) {
            int r = crow + v;
            if (r < M) C[(size_t)r * ldc + ccol] = acc[t][v];
            if (Chf && r < NPAD) {               // fragment-packed f16 copy (zero padded rows)
                int rr = r & 31;
                int lane2 = (lane & 15) + ((rr >> 4) << 4);
                _Float16 hx = (r < M) ? (_Float16)acc[t][v] : (_Float16)0.f;
                Chf[((size_t)(r >> 5) * nT + (col0 >> 4) + t) * 512 + lane2 * 16 + (rr & 15)] = hx;
            }
        }
    }
}

// ---------------- src/dst attention logits ----------------
__global__ void k_srcdst1(const float* __restrict__ Hc, const float* __restrict__ ah,
                          float* __restrict__ srcH, float* __restrict__ dstH) {
    int idx = blockIdx.x * 256 + threadIdx.x;    // h*NPAD + i
    if (idx >= HEADS * NPAD) return;
    int h = idx / NPAD, i = idx % NPAD;
    float s = 0.f, d = 0.f;
    if (i < NNODES) {
        const float* row = Hc + (size_t)i * F1 + h * HID;
        const float* a = ah + h * 2 * HID;
        for (int f = 0; f < HID; ++f) { s += row[f] * a[f]; d += row[f] * a[HID + f]; }
    }
    srcH[idx] = s; dstH[idx] = d;
}

__global__ void k_srcdst2(const float* __restrict__ H2, const float* __restrict__ ao,
                          float* __restrict__ src2, float* __restrict__ dst2) {
    int i = blockIdx.x * 256 + threadIdx.x;
    if (i >= NPAD) return;
    float s = 0.f, d = 0.f;
    if (i < NNODES)
        for (int c = 0; c < NCLS; ++c) {
            s += H2[(size_t)i * NCLS + c] * ao[c];
            d += H2[(size_t)i * NCLS + c] * ao[NCLS + c];
        }
    src2[i] = s; dst2[i] = d;
}

// -------- online row softmax stats (all NH heads in one adjacency pass) --------
template <int NH>
__global__ void k_stats(const float* __restrict__ A,
                        const float* __restrict__ src, const float* __restrict__ dst,
                        float* __restrict__ mOut, float* __restrict__ sOut) {
    int i = blockIdx.x;
    __shared__ float redm[256 * NH];
    __shared__ float reds[256 * NH];
    float mv[NH], sv[NH], srl[NH];
#pragma unroll
    for (int h = 0; h < NH; ++h) {
        mv[h] = -3.0e38f; sv[h] = 0.f;
        srl[h] = (i < NNODES) ? src[h * NPAD + i] : 0.f;
    }
    if (i < NNODES) {
        const float* Arow = A + (size_t)i * NNODES;
        for (int j = threadIdx.x; j < NNODES; j += 256) {
            __builtin_prefetch(Arow + j + 256, 0, 1);    // global_prefetch_b8
            float a1 = Arow[j] + ((j == i) ? 1.f : 0.f);
            if (a1 > 0.f) {
#pragma unroll
                for (int h = 0; h < NH; ++h) {
                    float e = srl[h] + dst[h * NPAD + j];
                    e = (e > 0.f) ? e : GALPHA * e;
                    float nm = fmaxf(mv[h], e);
                    sv[h] = sv[h] * __expf(mv[h] - nm) + __expf(e - nm);
                    mv[h] = nm;
                }
            }
        }
    }
#pragma unroll
    for (int h = 0; h < NH; ++h) { redm[h * 256 + threadIdx.x] = mv[h]; reds[h * 256 + threadIdx.x] = sv[h]; }
    __syncthreads();
    for (int st = 128; st > 0; st >>= 1) {
        if (threadIdx.x < st) {
#pragma unroll
            for (int h = 0; h < NH; ++h) {
                float m1 = redm[h * 256 + threadIdx.x], m2 = redm[h * 256 + threadIdx.x + st];
                float s1 = reds[h * 256 + threadIdx.x], s2 = reds[h * 256 + threadIdx.x + st];
                float m = fmaxf(m1, m2);
                redm[h * 256 + threadIdx.x] = m;
                reds[h * 256 + threadIdx.x] = s1 * __expf(m1 - m) + s2 * __expf(m2 - m);
            }
        }
        __syncthreads();
    }
    if (threadIdx.x == 0) {
#pragma unroll
        for (int h = 0; h < NH; ++h) {
            float m = redm[h * 256], s = reds[h * 256];
            if (i >= NNODES || s <= 0.f) { m = 0.f; s = 1.f; }
            mOut[h * NPAD + i] = m; sOut[h * NPAD + i] = s;
        }
    }
}

// ---------------- fused flash-style att @ H with WMMA ----------------
// grid.y = head; tile 128 rows x BN feats; K (neighbor) step 32.
// P tile built on-the-fly (f16, LDS); H read as B fragments straight from global frag
// buffer. Branch-free padding: clamped adjacency address; dR/dC multiplicatively zero
// padded rows/cols and the H fragments are zero-filled for padded K, so no NaN leaks.
template <int BN, bool DOUBLE_ELU>
__global__ void k_att(const float* __restrict__ A, const float* __restrict__ dinv,
                      const float* __restrict__ src, const float* __restrict__ dst,
                      const float* __restrict__ mRow, const float* __restrict__ sRow,
                      const _Float16* __restrict__ Hf, int nTH,
                      float* __restrict__ out, int ldo) {
    constexpr int AS = 40;
    __shared__ alignas(16) _Float16 Ps[128 * AS];
    __shared__ float srcR[128], mR[128], sR[128], dR[128];
    __shared__ float dstC[32], dC[32];
    int tid = threadIdx.x, wave = tid >> 5, lane = tid & 31;
    int h = blockIdx.y;
    int row0 = blockIdx.x * 128;
    int cbT = h * (BN >> 4);
    const float* srcv = src + h * NPAD;
    const float* dstv = dst + h * NPAD;
    const float* mv = mRow + h * NPAD;
    const float* sv = sRow + h * NPAD;

    if (tid < 128) {
        int r = row0 + tid;
        bool ok = r < NNODES;
        srcR[tid] = ok ? srcv[r] : 0.f;
        mR[tid]   = ok ? mv[r] : 0.f;
        sR[tid]   = 1.f / (ok ? sv[r] : 1.f);
        dR[tid]   = ok ? dinv[r] : 0.f;          // 0 kills padded rows
    }
    v8f acc[BN / 16];
#pragma unroll
    for (int t = 0; t < BN / 16; ++t) acc[t] = (v8f)0.f;

    for (int j0 = 0; j0 < NPAD; j0 += 32) {
        if (tid < 32) {
            int j = j0 + tid;
            bool ok = j < NNODES;
            dstC[tid] = ok ? dstv[j] : 0.f;
            dC[tid]   = ok ? dinv[j] : 0.f;      // 0 kills padded cols
        }
        __syncthreads();
        // build normalized probability tile P (128x32) in f16; float4 in, b64 out
#pragma unroll
        for (int u = 0; u < 4; ++u) {
            int q = tid + u * 256;
            int r = q >> 3, kk = (q & 7) * 4;
            int gi = min(row0 + r, NNODES - 1);            // clamp for address safety
            int gj = min(j0 + kk, NNODES - 4);
            const float* ap = A + (size_t)gi * NNODES + gj;
            __builtin_prefetch(ap + 32, 0, 1);
            float4 a4 = *(const float4*)ap;
            float sc = sR[r] * dR[r];
            v4h pv;
#pragma unroll
            for (int e = 0; e < 4; ++e) {
                float ae = (e == 0) ? a4.x : (e == 1) ? a4.y : (e == 2) ? a4.z : a4.w;
                float a1 = ae + ((gj + e == gi) ? 1.f : 0.f);
                float p = 0.f;
                if (a1 > 0.f) {
                    float ev = srcR[r] + dstC[kk + e];
                    ev = (ev > 0.f) ? ev : GALPHA * ev;
                    p = __expf(ev - mR[r]) * sc * dC[kk + e] * a1;
                }
                pv[e] = (_Float16)p;
            }
            *(v4h*)&Ps[r * AS + kk] = pv;
        }
        __syncthreads();

        int abase = (wave * 16 + (lane & 15)) * AS + ((lane >> 4) << 4);
        v16h af = joinh(*(const v8h*)&Ps[abase], *(const v8h*)&Ps[abase + 8]);

#pragma unroll
        for (int t = 0; t < BN / 16; ++t) {
            const _Float16* bp = Hf + ((size_t)(j0 >> 5) * nTH + cbT + t) * 512 + lane * 16;
            v16h bf = joinh(*(const v8h*)bp, *(const v8h*)(bp + 8));
            acc[t] = __builtin_amdgcn_wmma_f32_16x16x32_f16(
                false, af, false, bf, (short)0, acc[t], false, false);
        }
        __syncthreads();
    }

    int crow = row0 + wave * 16 + (lane >> 4) * 8;
#pragma unroll
    for (int t = 0; t < BN / 16; ++t) {
        int ccol = cbT * 16 + t * 16 + (lane & 15);
#pragma unroll
        for (int v = 0; v < 8; ++v) {
            int r = crow + v;
            if (r < NNODES) {
                float x = eluf(acc[t][v]);
                if (DOUBLE_ELU) x = eluf(x);
                out[(size_t)r * ldo + ccol] = x;
            }
        }
    }
}

// ---------------- FC1 -> FC2 -> log_softmax head ----------------
__global__ void k_head(const float* __restrict__ X2, const float* __restrict__ FC1,
                       const float* __restrict__ FC2, float* __restrict__ out) {
    int i = blockIdx.x * 256 + threadIdx.x;
    if (i >= NNODES) return;
    float x[NCLS], y[NCLS];
#pragma unroll
    for (int c = 0; c < NCLS; ++c) x[c] = X2[(size_t)i * NCLS + c];
#pragma unroll
    for (int c = 0; c < NCLS; ++c) {
        float s = 0.f;
#pragma unroll
        for (int k = 0; k < NCLS; ++k) s += x[k] * FC1[c * NCLS + k];
        y[c] = eluf(s);
    }
#pragma unroll
    for (int c = 0; c < NCLS; ++c) {
        float s = 0.f;
#pragma unroll
        for (int k = 0; k < NCLS; ++k) s += y[k] * FC2[c * NCLS + k];
        x[c] = eluf(s);
    }
    float m = -3.0e38f;
#pragma unroll
    for (int c = 0; c < NCLS; ++c) m = fmaxf(m, x[c]);
    float s = 0.f;
#pragma unroll
    for (int c = 0; c < NCLS; ++c) s += __expf(x[c] - m);
    float l = m + __logf(s);
#pragma unroll
    for (int c = 0; c < NCLS; ++c) out[(size_t)i * NCLS + c] = x[c] - l;
}

extern "C" void kernel_launch(void* const* d_in, const int* in_sizes, int n_in,
                              void* d_out, int out_size, void* d_ws, size_t ws_size,
                              hipStream_t stream) {
    const float* A   = (const float*)d_in[0];
    const float* X   = (const float*)d_in[1];
    const float* Wh  = (const float*)d_in[2];
    const float* ah  = (const float*)d_in[3];
    const float* Wo  = (const float*)d_in[4];
    const float* ao  = (const float*)d_in[5];
    const float* FC1 = (const float*)d_in[6];
    const float* FC2 = (const float*)d_in[7];
    float* out = (float*)d_out;

    char* wp = (char*)d_ws;
    auto alloc = [&](size_t bytes) -> void* {
        void* p = (void*)wp;
        wp += (bytes + 255) & ~(size_t)255;
        return p;
    };
    float*    dinv  = (float*)alloc(NPAD * 4);
    float*    srcH  = (float*)alloc((size_t)HEADS * NPAD * 4);
    float*    dstH  = (float*)alloc((size_t)HEADS * NPAD * 4);
    float*    mH    = (float*)alloc((size_t)HEADS * NPAD * 4);
    float*    sH    = (float*)alloc((size_t)HEADS * NPAD * 4);
    float*    Hc    = (float*)alloc((size_t)NNODES * F1 * 4);
    _Float16* Hcf   = (_Float16*)alloc((size_t)(NPAD / 32) * (F1 / 16) * 512 * 2);
    _Float16* Wcat  = (_Float16*)alloc((size_t)(INDIM / 32) * (F1 / 16) * 512 * 2);
    float*    X1    = (float*)alloc((size_t)NNODES * F1 * 4);
    float*    H2    = (float*)alloc((size_t)NNODES * NCLS * 4);
    _Float16* H2f   = (_Float16*)alloc((size_t)(NPAD / 32) * 512 * 2);
    _Float16* Woutf = (_Float16*)alloc((size_t)(INDIM / 32) * 512 * 2);
    float*    src2  = (float*)alloc(NPAD * 4);
    float*    dst2  = (float*)alloc(NPAD * 4);
    float*    m2    = (float*)alloc(NPAD * 4);
    float*    s2    = (float*)alloc(NPAD * 4);
    float*    X2    = (float*)alloc((size_t)NNODES * NCLS * 4);

    const int MB = (NNODES + 127) / 128;   // 24 row blocks

    k_degree<<<NPAD, 256, 0, stream>>>(A, dinv);
    k_wcat<<<((INDIM / 32) * (F1 / 16) * 512 + 255) / 256, 256, 0, stream>>>(Wh, Wcat);
    k_wout<<<((INDIM / 32) * 512 + 255) / 256, 256, 0, stream>>>(Wo, Woutf);

    // layer 1: H = X @ Wcat  (all heads in one GEMM)
    k_gemm<64><<<dim3(MB, F1 / 64), 256, 0, stream>>>(X, INDIM, Wcat, Hc, Hcf, F1,
                                                      NNODES, F1, INDIM);
    k_srcdst1<<<(HEADS * NPAD + 255) / 256, 256, 0, stream>>>(Hc, ah, srcH, dstH);
    k_stats<HEADS><<<NPAD, 256, 0, stream>>>(A, srcH, dstH, mH, sH);
    k_att<64, false><<<dim3(MB, HEADS), 256, 0, stream>>>(A, dinv, srcH, dstH, mH, sH,
                                                          Hcf, F1 / 16, X1, F1);

    // layer 2
    k_gemm<16><<<dim3(MB, 1), 256, 0, stream>>>(X1, F1, Woutf, H2, H2f, NCLS,
                                                NNODES, NCLS, INDIM);
    k_srcdst2<<<(NPAD + 255) / 256, 256, 0, stream>>>(H2, ao, src2, dst2);
    k_stats<1><<<NPAD, 256, 0, stream>>>(A, src2, dst2, m2, s2);
    k_att<16, true><<<dim3(MB, 1), 256, 0, stream>>>(A, dinv, src2, dst2, m2, s2,
                                                     H2f, 1, X2, NCLS);

    // head
    k_head<<<(NNODES + 255) / 256, 256, 0, stream>>>(X2, FC1, FC2, out);
}
